// SwinTransformerBlock_1640677507080
// MI455X (gfx1250) — matrix operand order repaired and checked by reading, not compiled, not run on hardware
//
#include <hip/hip_runtime.h>

// ---------------- problem constants ----------------
#define Bb   32
#define Hh   56
#define Ww_  56
#define Cc   192
#define NHh  6
#define WSw  7
#define SSs  3
#define Ll   (Hh*Ww_)          // 3136
#define Nn   (WSw*WSw)         // 49
#define HDd  32                // C/NH
#define NWIN 64                // (56/7)^2 windows per image
#define TOK  (Bb*Ll)           // 100352 rows, = 1568*64
#define SCALEQ 0.17677669529663687f  // 32^-0.5

typedef __attribute__((ext_vector_type(16))) _Float16 v16h;
typedef __attribute__((ext_vector_type(8)))  float    v8f;

// ---------------- CDNA5 async copy helpers ----------------
// LDS byte address = low 32 bits of the flat address of a __shared__ pointer
// (flat LDS aperture keeps the AS3 offset in addr[31:0]).
__device__ __forceinline__ unsigned lds_addr(const void* p) {
    return (unsigned)(uintptr_t)p;
}
// GLOBAL_LOAD_ASYNC_TO_LDS_B128, GVS mode: mem = SADDR + VADDR[31:0].
// Tracked by ASYNCcnt (CDNA5 ISA).
__device__ __forceinline__ void async_ld_b128(unsigned ldsOff, unsigned gByteOff,
                                              const void* base) {
    asm volatile("global_load_async_to_lds_b128 %0, %1, %2"
                 :: "v"(ldsOff), "v"(gByteOff), "s"(base) : "memory");
}
__device__ __forceinline__ void wait_asynccnt0() {
    asm volatile("s_wait_asynccnt 0x0" ::: "memory");
}

// =====================================================================
// f32 -> f16 weight conversion WITH transpose: dst[n*K + k] = src[k*N + n]
// (B^T layout makes WMMA B-fragment loads contiguous in LDS)
// =====================================================================
__global__ void cvt_f16_t_kernel(const float* __restrict__ src,
                                 _Float16* __restrict__ dst, int K, int N) {
    int i = blockIdx.x * 256 + threadIdx.x;
    if (i < K * N) {
        int k = i / N, n = i % N;
        dst[n * K + k] = (_Float16)src[i];
    }
}

// =====================================================================
// LayerNorm (one wave32 per token, 6 channels/lane).
// windowed=1: output rows are shifted+window-partitioned; 0: identity
// =====================================================================
__global__ __launch_bounds__(128)
void ln_kernel(const float* __restrict__ x, const float* __restrict__ gamma,
               const float* __restrict__ beta, _Float16* __restrict__ out,
               int windowed) {
    int token = blockIdx.x * 4 + (threadIdx.x >> 5);
    int lane  = threadIdx.x & 31;
    size_t srcRow, dstRow;
    if (windowed) {
        int win = token / Nn, n = token % Nn;
        int b = win >> 6, wi = win & 63;
        int wh = wi >> 3, ww = wi & 7;
        int i = n / WSw, j = n % WSw;
        int h = (wh * WSw + i + SSs) % Hh;   // roll(-SS) source coordinate
        int w = (ww * WSw + j + SSs) % Ww_;
        srcRow = (size_t)b * Ll + h * Ww_ + w;
        dstRow = token;
    } else {
        srcRow = dstRow = token;
    }
    const float* xr = x + srcRow * Cc;
    float vals[6];
    float s = 0.f, s2 = 0.f;
#pragma unroll
    for (int k = 0; k < 6; ++k) {
        float v = xr[lane + 32 * k];
        vals[k] = v; s += v; s2 += v * v;
    }
#pragma unroll
    for (int m = 16; m >= 1; m >>= 1) {
        s  += __shfl_xor(s,  m, 32);
        s2 += __shfl_xor(s2, m, 32);
    }
    float mean = s * (1.0f / Cc);
    float var  = s2 * (1.0f / Cc) - mean * mean;
    float rstd = rsqrtf(var + 1e-3f);
    _Float16* orow = out + dstRow * Cc;
#pragma unroll
    for (int k = 0; k < 6; ++k) {
        int c = lane + 32 * k;
        orow[c] = (_Float16)((vals[k] - mean) * rstd * gamma[c] + beta[c]);
    }
}

// =====================================================================
// Generic WMMA GEMM:  C(M x N) = A(M x KV, f16 rowmajor) * B(KV x N)
// with B supplied TRANSPOSED (Bt: N x KV rowmajor).  KV compile-time.
// 128 threads = 4 waves; block tile 64x64; wave tile 16x64; K-step 32.
// Double-buffered LDS tiles filled with async-to-LDS DMA (ASYNCcnt),
// fully unrolled so async fill of tile k+1 interleaves with WMMAs of k.
// =====================================================================
template <int KV, class Epi>
__global__ __launch_bounds__(128)
void gemm_wmma_kernel(const _Float16* __restrict__ A,
                      const _Float16* __restrict__ Bt,
                      Epi epi) {
    __shared__ alignas(16) _Float16 sAb[2][64 * 32];
    __shared__ alignas(16) _Float16 sBb[2][64 * 32];
    const int tid  = threadIdx.x;
    const int wave = tid >> 5;
    const int lane = tid & 31;
    const int m0 = blockIdx.x * 64;
    const int n0 = blockIdx.y * 64;

    // async fill of one K-tile (A: rows m0.., Bt: rows n0..), 64x32 f16 each
    auto fill = [&](int kk, _Float16* sA, _Float16* sB) {
        int idx = tid;
#pragma unroll
        for (int r = 0; r < 2; ++r, idx += 128) {
            int row = idx >> 2, seg = idx & 3;
            unsigned la = lds_addr(sA + row * 32 + seg * 8);
            unsigned ga = (unsigned)(((unsigned)(m0 + row) * (unsigned)KV + kk + seg * 8) * 2u);
            async_ld_b128(la, ga, A);
            unsigned lb = lds_addr(sB + row * 32 + seg * 8);
            unsigned gb = (unsigned)(((unsigned)(n0 + row) * (unsigned)KV + kk + seg * 8) * 2u);
            async_ld_b128(lb, gb, Bt);
        }
    };

    v8f acc[4] = {};

    fill(0, sAb[0], sBb[0]);
    wait_asynccnt0();
    __syncthreads();

#pragma unroll
    for (int ks = 0; ks < KV / 32; ++ks) {
        const int cur = ks & 1;
        if (ks + 1 < KV / 32) fill((ks + 1) * 32, sAb[cur ^ 1], sBb[cur ^ 1]);

        const _Float16* sA = sAb[cur];
        const _Float16* sB = sBb[cur];

        // A fragment: lane half selects K sub-ranges (0-15 | 16-31)
        v16h a;
        {
            const _Float16* ar = sA + (wave * 16 + (lane & 15)) * 32;
            int off = (lane < 16) ? 0 : 16;
            union { v16h v; uint4 q[2]; } ua;
            ua.q[0] = *(const uint4*)(ar + off);
            ua.q[1] = *(const uint4*)(ar + off + 8);
            a = ua.v;
        }
#pragma unroll
        for (int t = 0; t < 4; ++t) {
            // B fragment: lane n = column of B = row of Bt (contiguous K run)
            v16h b;
            {
                const _Float16* br = sB + (t * 16 + (lane & 15)) * 32;
                int kb = (lane < 16) ? 0 : 16;
                union { v16h v; uint4 q[2]; } ub;
                ub.q[0] = *(const uint4*)(br + kb);
                ub.q[1] = *(const uint4*)(br + kb + 8);
                b = ub.v;
            }
            acc[t] = __builtin_amdgcn_wmma_f32_16x16x32_f16(
                false, a, false, b, (short)0, acc[t], false, false);
        }
        wait_asynccnt0();
        __syncthreads();
    }

    // epilogue: D layout row = p + 8*(lane>=16), col = lane&15
#pragma unroll
    for (int t = 0; t < 4; ++t) {
        int n  = n0 + t * 16 + (lane & 15);
        int mB = m0 + wave * 16 + ((lane < 16) ? 0 : 8);
#pragma unroll
        for (int p = 0; p < 8; ++p) epi(mB + p, n, acc[t][p]);
    }
}

// ------------------------- epilogues -------------------------
struct EpiQKV {
    const float* bias;          // 576
    _Float16 *q, *k, *v;        // q,k: [2048*6][49][32]; v: [2048*6][32][64] (transposed, n-padded)
    __device__ void operator()(int r, int c, float acc) const {
        float val = acc + bias[c];
        int win = r / Nn, n = r % Nn;
        int s = c / Cc, rem = c % Cc;
        int head = rem >> 5, hd = rem & 31;
        size_t wh = (size_t)win * NHh + head;
        if (s == 0)      q[(wh * Nn + n) * HDd + hd] = (_Float16)(val * SCALEQ);
        else if (s == 1) k[(wh * Nn + n) * HDd + hd] = (_Float16)val;
        else             v[(wh * HDd + hd) * 64 + n] = (_Float16)val;  // V^T
    }
};

struct EpiProj {
    const float* bias; const float* xin; float* out;
    __device__ void operator()(int r, int c, float acc) const {
        float val = acc + bias[c];
        int win = r / Nn, n = r % Nn;
        int b = win >> 6, wi = win & 63;
        int wh = wi >> 3, ww = wi & 7;
        int i = n / WSw, j = n % WSw;
        int h = (wh * WSw + i + SSs) % Hh;   // window reverse + roll(+SS)
        int w = (ww * WSw + j + SSs) % Ww_;
        size_t tok = (size_t)b * Ll + h * Ww_ + w;
        out[tok * Cc + c] = xin[tok * Cc + c] + val;   // shortcut + attn
    }
};

struct EpiFC1 {
    const float* bias; _Float16* hbuf;
    __device__ void operator()(int r, int c, float acc) const {
        float x = acc + bias[c];
        float g = 0.5f * x * (1.0f + erff(x * 0.7071067811865475f)); // exact gelu
        hbuf[(size_t)r * (4 * Cc) + c] = (_Float16)g;
    }
};

struct EpiFC2 {
    const float* bias; float* out;
    __device__ void operator()(int r, int c, float acc) const {
        out[(size_t)r * Cc + c] += acc + bias[c];      // x + mlp(y)
    }
};

// =====================================================================
// Attention: one block (4 waves) per (window, head). N=49 padded to 64.
//   S = Q K^T (+bias+mask) -> softmax -> O = P V
// =====================================================================
__device__ __forceinline__ int regionOf(int h) {
    return (h < Hh - WSw) ? 0 : (h < Hh - SSs) ? 1 : 2;
}

__global__ __launch_bounds__(128)
void attn_kernel(const _Float16* __restrict__ Q, const _Float16* __restrict__ Km,
                 const _Float16* __restrict__ Vt, const float* __restrict__ relTab,
                 _Float16* __restrict__ O) {
    __shared__ alignas(16) _Float16 sQ[64 * 32];
    __shared__ alignas(16) _Float16 sK[64 * 32];
    __shared__ alignas(16) _Float16 sVt[32 * 64];
    __shared__ alignas(16) _Float16 sP[64 * 64];

    const int wh_idx = blockIdx.x;           // win*NH + head
    const int win  = wh_idx / NHh;
    const int head = wh_idx % NHh;
    const int tid  = threadIdx.x;
    const int wave = tid >> 5;
    const int lane = tid & 31;

    const _Float16* qb = Q  + (size_t)wh_idx * Nn * HDd;
    const _Float16* kb = Km + (size_t)wh_idx * Nn * HDd;
    const _Float16* vb = Vt + (size_t)wh_idx * HDd * 64;

    // ---- fill Q/K (rows 49..63 zero) via async DMA; V^T with zero pad cols ----
#pragma unroll
    for (int mat = 0; mat < 2; ++mat) {                 // uniform matrix select
        const _Float16* src = mat ? kb : qb;
        _Float16* dst = mat ? sK : sQ;
        for (int local = tid; local < 256; local += 128) {
            int row = local >> 2, seg = local & 3;
            if (row < Nn) {
                async_ld_b128(lds_addr(dst + row * 32 + seg * 8),
                              (unsigned)((row * 32 + seg * 8) * 2), src);
            } else {
                uint4 z = {};
                *((uint4*)(dst + row * 32) + seg) = z;
            }
        }
    }
    for (int local = tid; local < 256; local += 128) {   // V^T: 32 rows x 64 cols
        int row = local >> 3, seg = local & 7;
        if (seg < 6) {
            async_ld_b128(lds_addr(sVt + row * 64 + seg * 8),
                          (unsigned)((row * 64 + seg * 8) * 2), vb);
        } else {
            union { uint4 q; _Float16 h[8]; } tz;
#pragma unroll
            for (int e = 0; e < 8; ++e) {
                int n = seg * 8 + e;
                tz.h[e] = (n < Nn) ? vb[row * 64 + n] : (_Float16)0;
            }
            *((uint4*)(sVt + row * 64) + seg) = tz.q;
        }
    }
    wait_asynccnt0();
    __syncthreads();

    // ---- S = Q K^T : wave handles rows [16w,16w+16), K-dim = 32 (1 WMMA/tile) ----
    v16h aq;
    {
        const _Float16* ar = sQ + (wave * 16 + (lane & 15)) * HDd;
        int off = (lane < 16) ? 0 : 16;
        union { v16h v; uint4 q[2]; } ua;
        ua.q[0] = *(const uint4*)(ar + off);
        ua.q[1] = *(const uint4*)(ar + off + 8);
        aq = ua.v;
    }
    v8f zacc = {};
    v8f sacc[4];
#pragma unroll
    for (int t = 0; t < 4; ++t) {
        // B = K^T: column n of K^T is row n of K (contiguous in LDS)
        v16h b;
        {
            const _Float16* krow = sK + (t * 16 + (lane & 15)) * HDd;
            int kbase = (lane < 16) ? 0 : 16;
            union { v16h v; uint4 q[2]; } ub;
            ub.q[0] = *(const uint4*)(krow + kbase);
            ub.q[1] = *(const uint4*)(krow + kbase + 8);
            b = ub.v;
        }
        sacc[t] = __builtin_amdgcn_wmma_f32_16x16x32_f16(
            false, aq, false, b, (short)0, zacc, false, false);
    }

    // ---- bias + shift mask + padding ----
    const int wIdx = win & 63;
    const int whp = wIdx >> 3, wwp = wIdx & 7;
    float sc[4][8];
#pragma unroll
    for (int t = 0; t < 4; ++t) {
#pragma unroll
        for (int p = 0; p < 8; ++p) {
            int m  = wave * 16 + p + ((lane < 16) ? 0 : 8);   // query idx
            int nk = t * 16 + (lane & 15);                     // key idx
            float val;
            if (nk < Nn) {
                if (m < Nn) {
                    int i1 = m / WSw,  j1 = m % WSw;
                    int i2 = nk / WSw, j2 = nk % WSw;
                    int rel = (i1 - i2 + WSw - 1) * (2 * WSw - 1) + (j1 - j2 + WSw - 1);
                    float bias = relTab[rel * NHh + head];
                    int id1 = regionOf(whp * WSw + i1) * 3 + regionOf(wwp * WSw + j1);
                    int id2 = regionOf(whp * WSw + i2) * 3 + regionOf(wwp * WSw + j2);
                    float msk = (id1 != id2) ? -100.0f : 0.0f;
                    val = sacc[t][p] + bias + msk;
                } else {
                    val = 0.0f;                 // padded query row: keep finite
                }
            } else {
                val = -1e30f;                    // padded key col: exp -> 0
            }
            sc[t][p] = val;
        }
    }

    // ---- softmax over 64 cols (4 tiles x 16-lane half-wave) ----
#pragma unroll
    for (int p = 0; p < 8; ++p) {
        float mx = -1e30f;
#pragma unroll
        for (int t = 0; t < 4; ++t) mx = fmaxf(mx, sc[t][p]);
#pragma unroll
        for (int msk = 8; msk >= 1; msk >>= 1) mx = fmaxf(mx, __shfl_xor(mx, msk, 32));
        float e[4]; float sum = 0.f;
#pragma unroll
        for (int t = 0; t < 4; ++t) { e[t] = __expf(sc[t][p] - mx); sum += e[t]; }
#pragma unroll
        for (int msk = 8; msk >= 1; msk >>= 1) sum += __shfl_xor(sum, msk, 32);
        float inv = 1.0f / sum;
        int m = wave * 16 + p + ((lane < 16) ? 0 : 8);
#pragma unroll
        for (int t = 0; t < 4; ++t)
            sP[m * 64 + t * 16 + (lane & 15)] = (_Float16)(e[t] * inv);
    }
    __syncthreads();

    // ---- O = P V : 16x32 per wave, K = 64 (two K-steps); B from V^T ----
    v8f oacc[2] = {};
#pragma unroll
    for (int kc = 0; kc < 64; kc += 32) {
        v16h a;
        {
            const _Float16* ar = sP + (wave * 16 + (lane & 15)) * 64 + kc;
            int off = (lane < 16) ? 0 : 16;
            union { v16h v; uint4 q[2]; } ua;
            ua.q[0] = *(const uint4*)(ar + off);
            ua.q[1] = *(const uint4*)(ar + off + 8);
            a = ua.v;
        }
#pragma unroll
        for (int t = 0; t < 2; ++t) {
            v16h b;
            {
                const _Float16* vr = sVt + (t * 16 + (lane & 15)) * 64 + kc;
                int kbase = (lane < 16) ? 0 : 16;
                union { v16h v; uint4 q[2]; } ub;
                ub.q[0] = *(const uint4*)(vr + kbase);
                ub.q[1] = *(const uint4*)(vr + kbase + 8);
                b = ub.v;
            }
            oacc[t] = __builtin_amdgcn_wmma_f32_16x16x32_f16(
                false, a, false, b, (short)0, oacc[t], false, false);
        }
    }

    // ---- store O (merge heads): row = win*49+m, col = head*32 + n ----
#pragma unroll
    for (int t = 0; t < 2; ++t) {
#pragma unroll
        for (int p = 0; p < 8; ++p) {
            int m = wave * 16 + p + ((lane < 16) ? 0 : 8);
            if (m < Nn) {
                size_t o = ((size_t)win * Nn + m) * Cc + head * HDd + t * 16 + (lane & 15);
                O[o] = (_Float16)oacc[t][p];
            }
        }
    }
}

// =====================================================================
// launcher
// =====================================================================
extern "C" void kernel_launch(void* const* d_in, const int* in_sizes, int n_in,
                              void* d_out, int out_size, void* d_ws, size_t ws_size,
                              hipStream_t stream) {
    const float* x      = (const float*)d_in[0];
    const float* gamma1 = (const float*)d_in[1];
    const float* beta1  = (const float*)d_in[2];
    const float* qkv_w  = (const float*)d_in[3];
    const float* qkv_b  = (const float*)d_in[4];
    const float* proj_w = (const float*)d_in[5];
    const float* proj_b = (const float*)d_in[6];
    const float* relTab = (const float*)d_in[7];
    const float* gamma2 = (const float*)d_in[8];
    const float* beta2  = (const float*)d_in[9];
    const float* fc1_w  = (const float*)d_in[10];
    const float* fc1_b  = (const float*)d_in[11];
    const float* fc2_w  = (const float*)d_in[12];
    const float* fc2_b  = (const float*)d_in[13];
    float* out = (float*)d_out;
    char*  ws  = (char*)d_ws;

    // workspace layout (bytes)
    constexpr size_t SZ_TOKC = (size_t)TOK * Cc * sizeof(_Float16);       // 38,535,168
    constexpr size_t SZ_VT   = (size_t)Bb * NWIN * NHh * HDd * 64 * 2;    // 50,331,648
    _Float16* WQt = (_Float16*)(ws);                         // qkv_w^T  (576 x 192)
    _Float16* WPt = (_Float16*)(ws + 221184ull);             // proj_w^T (192 x 192)
    _Float16* W1t = (_Float16*)(ws + 294912ull);             // fc1_w^T  (768 x 192)
    _Float16* W2t = (_Float16*)(ws + 589824ull);             // fc2_w^T  (192 x 768)
    _Float16* X16 = (_Float16*)(ws + 884736ull);
    _Float16* Q16 = (_Float16*)(ws + 884736ull + SZ_TOKC);
    _Float16* K16 = (_Float16*)(ws + 884736ull + 2 * SZ_TOKC);
    _Float16* V16 = (_Float16*)(ws + 884736ull + 3 * SZ_TOKC);            // V^T, padded
    _Float16* O16 = (_Float16*)(ws + 884736ull + 3 * SZ_TOKC + SZ_VT);
    _Float16* Y16 = X16;   // reuse: X16 dead after QKV GEMM
    _Float16* H16 = Q16;   // reuse: Q/K/V/O region (165.9MB >= 154.1MB) dead after proj

    // 1) weights -> f16, transposed
    cvt_f16_t_kernel<<<(110592 + 255) / 256, 256, 0, stream>>>(qkv_w, WQt, Cc, 3 * Cc);
    cvt_f16_t_kernel<<<(36864  + 255) / 256, 256, 0, stream>>>(proj_w, WPt, Cc, Cc);
    cvt_f16_t_kernel<<<(147456 + 255) / 256, 256, 0, stream>>>(fc1_w, W1t, Cc, 4 * Cc);
    cvt_f16_t_kernel<<<(147456 + 255) / 256, 256, 0, stream>>>(fc2_w, W2t, 4 * Cc, Cc);

    // 2) LN1 + shift + window partition -> X16
    ln_kernel<<<TOK / 4, 128, 0, stream>>>(x, gamma1, beta1, X16, 1);

    // 3) QKV GEMM (100352 x 192 x 576)
    {
        dim3 g(TOK / 64, (3 * Cc) / 64);
        gemm_wmma_kernel<Cc, EpiQKV><<<g, 128, 0, stream>>>(
            X16, WQt, EpiQKV{qkv_b, Q16, K16, V16});
    }

    // 4) windowed attention (12288 blocks)
    attn_kernel<<<Bb * NWIN * NHh, 128, 0, stream>>>(Q16, K16, V16, relTab, O16);

    // 5) proj GEMM + window reverse + unshift + residual -> d_out
    {
        dim3 g(TOK / 64, Cc / 64);
        gemm_wmma_kernel<Cc, EpiProj><<<g, 128, 0, stream>>>(
            O16, WPt, EpiProj{proj_b, x, out});
    }

    // 6) LN2 -> Y16
    ln_kernel<<<TOK / 4, 128, 0, stream>>>(out, gamma2, beta2, Y16, 0);

    // 7) FC1 + gelu -> H16
    {
        dim3 g(TOK / 64, (4 * Cc) / 64);
        gemm_wmma_kernel<Cc, EpiFC1><<<g, 128, 0, stream>>>(
            Y16, W1t, EpiFC1{fc1_b, H16});
    }

    // 8) FC2 + residual accumulate into d_out
    {
        dim3 g(TOK / 64, Cc / 64);
        gemm_wmma_kernel<4 * Cc, EpiFC2><<<g, 128, 0, stream>>>(
            H16, W2t, EpiFC2{fc2_b, out});
    }
}
